// NormLocalAttention_71871982731659
// MI455X (gfx1250) — matrix operand order repaired and checked by reading, not compiled, not run on hardware
//
#include <hip/hip_runtime.h>
#include <hip/hip_bf16.h>

typedef __attribute__((ext_vector_type(16))) _Float16 v16h;
typedef __attribute__((ext_vector_type(8)))  _Float16 v8h;
typedef __attribute__((ext_vector_type(8)))  float    v8f;
typedef __attribute__((__vector_size__(16))) int      v4i_t;

#define WMMA16(a, b, c) \
  __builtin_amdgcn_wmma_f32_16x16x32_f16(false, (a), false, (b), (short)0, (c), false, false)

static __device__ __forceinline__ v8f vzero8() {
  v8f z;
#pragma unroll
  for (int i = 0; i < 8; ++i) z[i] = 0.0f;
  return z;
}

// ---------------------------------------------------------------------------
// gfx1250 async global->LDS copy (ASYNCcnt path), with inline-asm fallback.
// Builtin parameters are addrspace-qualified pointers to 4xi32 vectors.
// ---------------------------------------------------------------------------
static __device__ __forceinline__ void async_b128(const _Float16* g, _Float16* lds) {
#if __has_builtin(__builtin_amdgcn_global_load_async_to_lds_b128)
  __builtin_amdgcn_global_load_async_to_lds_b128(
      (__attribute__((address_space(1))) v4i_t*)(g),
      (__attribute__((address_space(3))) v4i_t*)(lds), 0, 0);
#else
  uint32_t l = (uint32_t)(uintptr_t)((__attribute__((address_space(3))) _Float16*)lds);
  asm volatile("global_load_async_to_lds_b128 %0, %1, off"
               :: "v"(l), "v"(g) : "memory");
#endif
}

static __device__ __forceinline__ void wait_async0() {
#if __has_builtin(__builtin_amdgcn_s_wait_asynccnt)
  __builtin_amdgcn_s_wait_asynccnt(0);
#else
  asm volatile("s_wait_asynccnt 0x0" ::: "memory");
#endif
}

// A-matrix fragment (16x32 f16), source row-major with leading dim `ld`.
// Lane r = lane&15 holds row (row0+r); contiguous K run [kh*16,+16).
static __device__ __forceinline__ v16h frag_a(const _Float16* p, int ld, int row0, int k0, int lane) {
  const int r = lane & 15, kh = lane >> 4;
  const _Float16* q = p + (size_t)(row0 + r) * ld + (k0 + kh * 16);
  v8h lo = *(const v8h*)(q);
  v8h hi = *(const v8h*)(q + 8);
  v16h out;
#pragma unroll
  for (int i = 0; i < 8; ++i) { out[i] = lo[i]; out[i + 8] = hi[i]; }
  return out;
}

// B-matrix fragment (32x16 f16) for D = A x B with B[k][n] = W[n][k],
// W stored row-major (N rows, K contiguous) with leading dim `ld`.
// Lane n = lane&15 holds column (col0+n); contiguous K run [kh*16,+16).
static __device__ __forceinline__ v16h frag_b(const _Float16* p, int ld, int col0, int k0, int lane) {
  const int n = lane & 15, kh = lane >> 4;
  const _Float16* q = p + (size_t)(col0 + n) * ld + (k0 + kh * 16);
  return *(const v16h*)(q);
}

// ---------------------------------------------------------------------------
// fp32 -> f16 convert (optionally scaled)
// ---------------------------------------------------------------------------
__global__ void cvt_f16(const float* __restrict__ src, _Float16* __restrict__ dst,
                        int n, float scale) {
  int i = blockIdx.x * blockDim.x + threadIdx.x;
  if (i < n) dst[i] = (_Float16)(src[i] * scale);
}

// ---------------------------------------------------------------------------
// C = A @ W^T (+ bias*bscale).  A: MxK f16 row-major, W: NxK f16 row-major.
// Block = 256 threads = 8 waves; block tile 128(M) x 128(N); K staged through
// LDS in 64-wide stages, double-buffered with async global->LDS copies.
// Wave tile 32x64 (waves arranged 4 x 2).
// ---------------------------------------------------------------------------
template <bool HALF_OUT>
__global__ __launch_bounds__(256) void gemm_xwT(const _Float16* __restrict__ A,
                                                const _Float16* __restrict__ W,
                                                const float* __restrict__ bias, float bscale,
                                                float* __restrict__ Cf, _Float16* __restrict__ Ch,
                                                int M, int N, int K) {
  const int lane   = threadIdx.x & 31;
  const int wave   = threadIdx.x >> 5;
  const int wm     = (wave & 3) * 32;   // wave M offset in block tile
  const int wn     = (wave >> 2) * 64;  // wave N offset in block tile
  const int m_blk  = blockIdx.x * 128;
  const int n_blk  = blockIdx.y * 128;

  __shared__ __align__(32) _Float16 s_a[2][128 * 64];
  __shared__ __align__(32) _Float16 s_b[2][128 * 64];

  v8f acc[2][4];
#pragma unroll
  for (int i = 0; i < 2; ++i)
#pragma unroll
    for (int j = 0; j < 4; ++j) acc[i][j] = vzero8();

  // issue one 128x64 A-tile + 128x64 B-tile stage: 2048 b128 copies, 8/thread
  auto issue_stage = [&](int buf, int k0) {
#pragma unroll
    for (int i = 0; i < 4; ++i) {
      const int c   = threadIdx.x + i * 256;  // 0..1023
      const int row = c >> 3;                 // 0..127
      const int cc  = (c & 7) * 8;            // half offset within 64-wide row
      async_b128(A + (size_t)(m_blk + row) * K + k0 + cc, &s_a[buf][row * 64 + cc]);
      async_b128(W + (size_t)(n_blk + row) * K + k0 + cc, &s_b[buf][row * 64 + cc]);
    }
  };

  issue_stage(0, 0);
  wait_async0();
  __syncthreads();

  const int nstages = K >> 6;  // K/64
  for (int kt = 0; kt < nstages; ++kt) {
    const int buf = kt & 1;
    if (kt + 1 < nstages) issue_stage(buf ^ 1, (kt + 1) * 64);

    const _Float16* pa = s_a[buf];
    const _Float16* pb = s_b[buf];
#pragma unroll
    for (int ks = 0; ks < 64; ks += 32) {
      v16h a0 = frag_a(pa, 64, wm,      ks, lane);
      v16h a1 = frag_a(pa, 64, wm + 16, ks, lane);
      v16h b0 = frag_b(pb, 64, wn,      ks, lane);
      v16h b1 = frag_b(pb, 64, wn + 16, ks, lane);
      v16h b2 = frag_b(pb, 64, wn + 32, ks, lane);
      v16h b3 = frag_b(pb, 64, wn + 48, ks, lane);
      acc[0][0] = WMMA16(a0, b0, acc[0][0]);
      acc[0][1] = WMMA16(a0, b1, acc[0][1]);
      acc[0][2] = WMMA16(a0, b2, acc[0][2]);
      acc[0][3] = WMMA16(a0, b3, acc[0][3]);
      acc[1][0] = WMMA16(a1, b0, acc[1][0]);
      acc[1][1] = WMMA16(a1, b1, acc[1][1]);
      acc[1][2] = WMMA16(a1, b2, acc[1][2]);
      acc[1][3] = WMMA16(a1, b3, acc[1][3]);
    }

    // Drain own async copies (ASYNCcnt is per-wave) BEFORE the barrier so the
    // next stage's LDS tile is globally visible; barrier also protects WAR on
    // the buffer we are about to overwrite next iteration.
    wait_async0();
    __syncthreads();
  }

  const int n_in = lane & 15;
  const int mh   = (lane >> 4) * 8;
#pragma unroll
  for (int mi = 0; mi < 2; ++mi)
#pragma unroll
    for (int nj = 0; nj < 4; ++nj) {
      const int col = n_blk + wn + nj * 16 + n_in;
      const float bv = bias[col] * bscale;
#pragma unroll
      for (int jj = 0; jj < 8; ++jj) {
        const int row = m_blk + wm + mi * 16 + mh + jj;
        const float v = acc[mi][nj][jj] + bv;
        if (HALF_OUT) Ch[(size_t)row * N + col] = (_Float16)v;
        else          Cf[(size_t)row * N + col] = v;
      }
    }
}

// ---------------------------------------------------------------------------
// Local banded attention: one block per (window w, head h).
// dots = bq(64xd) . bk(192xd)^T ; relu + band mask ; out = attn(64x192) . bv(192x64)
// Block = 128 threads = 4 waves; wave owns a 16-row strip.
// ---------------------------------------------------------------------------
__global__ __launch_bounds__(128) void local_attn(const _Float16* __restrict__ qp,
                                                  const _Float16* __restrict__ kp,
                                                  const _Float16* __restrict__ vp,
                                                  float* __restrict__ ao) {
  const int w    = blockIdx.x;   // 0..127 windows
  const int h    = blockIdx.y;   // 0..15 heads
  const int lane = threadIdx.x & 31;
  const int wave = threadIdx.x >> 5;  // 0..3
  const int m0   = wave * 16;

  __shared__ __align__(32) _Float16 s_attn[64 * 208];  // attn (f16), ld=208
  __shared__ __align__(32) _Float16 s_vT[64 * 192];    // bv^T: [e][j], ld=192

  // Stage bv transposed into LDS (zero for out-of-range neighbor chunks).
  for (int idx = threadIdx.x; idx < 3 * 64 * 64; idx += 128) {
    const int b  = idx >> 12;
    const int rm = idx & 4095;
    const int j  = rm >> 6;
    const int e  = rm & 63;
    const int sw = w - 1 + b;
    _Float16 val = (_Float16)0.0f;
    if (sw >= 0 && sw < 128)
      val = vp[(size_t)(sw * 64 + j) * 1024 + h * 64 + e];
    s_vT[e * 192 + b * 64 + j] = val;
  }

  // dots + relu -> s_attn.  q rows are this wave's 16 rows; K = head dim (64).
  v16h aq0 = frag_a(qp, 1024, w * 64 + m0, h * 64 + 0,  lane);
  v16h aq1 = frag_a(qp, 1024, w * 64 + m0, h * 64 + 32, lane);
  const int n_in = lane & 15;
  const int mh   = (lane >> 4) * 8;

  for (int b = 0; b < 3; ++b) {          // neighbor chunk (block-uniform branch)
    const int sw = w - 1 + b;
    if (sw >= 0 && sw < 128) {
#pragma unroll
      for (int nt = 0; nt < 4; ++nt) {
        v8f acc = vzero8();
        v16h bk0 = frag_b(kp, 1024, sw * 64 + nt * 16, h * 64 + 0,  lane);
        v16h bk1 = frag_b(kp, 1024, sw * 64 + nt * 16, h * 64 + 32, lane);
        acc = WMMA16(aq0, bk0, acc);
        acc = WMMA16(aq1, bk1, acc);
        const int col = b * 64 + nt * 16 + n_in;
#pragma unroll
        for (int jj = 0; jj < 8; ++jj) {
          const float v = acc[jj];
          s_attn[(m0 + mh + jj) * 208 + col] = (_Float16)(v > 0.0f ? v : 0.0f);
        }
      }
    } else {
      // masked chunk: zero this wave's 16x64 attn region
      for (int jj = 0; jj < 16; ++jj)
        for (int c = lane; c < 64; c += 32)
          s_attn[(m0 + jj) * 208 + b * 64 + c] = (_Float16)0.0f;
    }
  }
  __syncthreads();

  // out = attn @ bv : A = s_attn (rows m0..m0+15, K=192), B = s_vT (N=64, K contig)
  v16h af[6];
#pragma unroll
  for (int ks = 0; ks < 6; ++ks) af[ks] = frag_a(s_attn, 208, m0, ks * 32, lane);

#pragma unroll
  for (int nt = 0; nt < 4; ++nt) {
    v8f acc = vzero8();
#pragma unroll
    for (int ks = 0; ks < 6; ++ks) {
      v16h bb = frag_b(s_vT, 192, nt * 16, ks * 32, lane);
      acc = WMMA16(af[ks], bb, acc);
    }
    const int col = h * 64 + nt * 16 + n_in;
#pragma unroll
    for (int jj = 0; jj < 8; ++jj)
      ao[(size_t)(w * 64 + m0 + mh + jj) * 1024 + col] = acc[jj];
  }
}

// ---------------------------------------------------------------------------
// Gated RMSNorm: y = scale * (x / (rms + eps)) * sigmoid(gate * x), f16 out.
// One block (256 threads) per row of 1024.
// ---------------------------------------------------------------------------
__global__ __launch_bounds__(256) void gated_rmsnorm(const float* __restrict__ x,
                                                     const float* __restrict__ scale,
                                                     const float* __restrict__ gate,
                                                     _Float16* __restrict__ y) {
  const int row = blockIdx.x;
  const float* xr = x + (size_t)row * 1024;
  __shared__ float red[256];
  float ss = 0.0f;
  for (int e = threadIdx.x; e < 1024; e += 256) { float v = xr[e]; ss += v * v; }
  red[threadIdx.x] = ss;
  __syncthreads();
  for (int s = 128; s > 0; s >>= 1) {
    if (threadIdx.x < s) red[threadIdx.x] += red[threadIdx.x + s];
    __syncthreads();
  }
  const float rms = sqrtf(red[0] * (1.0f / 1024.0f));
  const float inv = 1.0f / (rms + 1e-8f);
  for (int e = threadIdx.x; e < 1024; e += 256) {
    const float v = xr[e];
    const float sg = 1.0f / (1.0f + expf(-gate[e] * v));
    y[(size_t)row * 1024 + e] = (_Float16)(scale[e] * v * inv * sg);
  }
}

// ---------------------------------------------------------------------------
extern "C" void kernel_launch(void* const* d_in, const int* in_sizes, int n_in,
                              void* d_out, int out_size, void* d_ws, size_t ws_size,
                              hipStream_t stream) {
  const float* query  = (const float*)d_in[0];
  const float* key_   = (const float*)d_in[1];
  const float* value  = (const float*)d_in[2];
  const float* w_q    = (const float*)d_in[3];
  const float* b_q    = (const float*)d_in[4];
  const float* w_k    = (const float*)d_in[5];
  const float* b_k    = (const float*)d_in[6];
  const float* w_v    = (const float*)d_in[7];
  const float* b_v    = (const float*)d_in[8];
  const float* w_o    = (const float*)d_in[9];
  const float* b_o    = (const float*)d_in[10];
  const float* nscale = (const float*)d_in[11];
  const float* ngate  = (const float*)d_in[12];

  const int L = 8192, E = 1024;
  const size_t LE = (size_t)L * E;  // 8388608
  const size_t EE = (size_t)E * E;  // 1048576
  const float qscale = 0.125f;      // d^-0.5, d = 64 (folded into w_q / b_q)

  // Workspace layout (~152 MB)
  _Float16* xh = (_Float16*)d_ws;        // 3*LE f16: query/key/value in f16
  _Float16* wh = xh + 3 * LE;            // 4*EE f16: w_q(scaled), w_k, w_v, w_o
  _Float16* pp = wh + 4 * EE;            // 3*LE f16: q/k/v projections
  float*    ao = (float*)(pp + 3 * LE);  // LE f32: attention output
  _Float16* nh = (_Float16*)(ao + LE);   // LE f16: normed

  // fp32 -> f16 conversions
  cvt_f16<<<(int)(LE / 256), 256, 0, stream>>>(query, xh,          (int)LE, 1.0f);
  cvt_f16<<<(int)(LE / 256), 256, 0, stream>>>(key_,  xh + LE,     (int)LE, 1.0f);
  cvt_f16<<<(int)(LE / 256), 256, 0, stream>>>(value, xh + 2 * LE, (int)LE, 1.0f);
  cvt_f16<<<(int)(EE / 256), 256, 0, stream>>>(w_q, wh,          (int)EE, qscale);
  cvt_f16<<<(int)(EE / 256), 256, 0, stream>>>(w_k, wh + EE,     (int)EE, 1.0f);
  cvt_f16<<<(int)(EE / 256), 256, 0, stream>>>(w_v, wh + 2 * EE, (int)EE, 1.0f);
  cvt_f16<<<(int)(EE / 256), 256, 0, stream>>>(w_o, wh + 3 * EE, (int)EE, 1.0f);

  // q/k/v projections (x @ W^T + b), f16 outputs
  dim3 gg(64, 8), gb(256);  // (M/128, N/128)
  gemm_xwT<true><<<gg, gb, 0, stream>>>(xh,          wh,          b_q, qscale,
                                        nullptr, pp,          L, E, E);
  gemm_xwT<true><<<gg, gb, 0, stream>>>(xh + LE,     wh + EE,     b_k, 1.0f,
                                        nullptr, pp + LE,     L, E, E);
  gemm_xwT<true><<<gg, gb, 0, stream>>>(xh + 2 * LE, wh + 2 * EE, b_v, 1.0f,
                                        nullptr, pp + 2 * LE, L, E, E);

  // banded local attention (relu-kernelized, no softmax)
  local_attn<<<dim3(128, 16), 128, 0, stream>>>(pp, pp + LE, pp + 2 * LE, ao);

  // gated RMSNorm
  gated_rmsnorm<<<L, 256, 0, stream>>>(ao, nscale, ngate, nh);

  // output projection, f32 out + b_o
  gemm_xwT<false><<<gg, gb, 0, stream>>>(nh, wh + 3 * EE, b_o, 1.0f,
                                         (float*)d_out, nullptr, L, E, E);
}
